// SpatialTemporalAtomEncoder_26697516712053
// MI455X (gfx1250) — matrix operand order repaired and checked by reading, not compiled, not run on hardware
//
#include <hip/hip_runtime.h>
#include <hip/hip_bf16.h>
#include <stdint.h>

#define NTHREADS 128

constexpr int T_FRAMES = 16;
constexpr int M_ATOMS  = 4096;
constexpr int DIM      = 128;
constexpr int NHEADS   = 4;
constexpr int DHEAD    = 32;
constexpr int WIN      = 32;
constexpr int HKEYS    = 128;
constexpr int LAYERS   = 3;
constexpr int NWIN     = M_ATOMS / WIN;   // 128
constexpr int BT       = T_FRAMES;        // effective batch (B0 == 1)
constexpr int NTOK     = 1024;
constexpr int TS2      = 768;             // 2 * token_s
constexpr int HID2     = 256;             // 2 * D
constexpr int OFFK     = (HKEYS - WIN) / 2;  // 48
constexpr float INV_SQRT_DH = 0.17677669529663687f;

typedef __attribute__((ext_vector_type(16))) __bf16 bf16x16;
typedef __attribute__((ext_vector_type(8)))  float  f32x8;

union FragAB { bf16x16 v; uint32_t u[8]; };
union FragC  { f32x8   v; float    f[8]; };

// ---------------- scalar helpers ----------------
__device__ __forceinline__ uint16_t f2bf(float f) {
  union { float f; uint32_t u; } x; x.f = f;
  uint32_t r;
  if ((x.u & 0x7F800000u) == 0x7F800000u) r = x.u;          // Inf/NaN passthrough
  else r = x.u + 0x7FFFu + ((x.u >> 16) & 1u);              // RNE
  return (uint16_t)(r >> 16);
}
__device__ __forceinline__ float bf2f(uint16_t b) {
  union { uint32_t u; float f; } x; x.u = ((uint32_t)b) << 16; return x.f;
}
__device__ __forceinline__ float sigm(float x) { return 1.0f / (1.0f + __expf(-x)); }
__device__ __forceinline__ float softplusf(float x) { return (x > 20.f) ? x : log1pf(__expf(x)); }

__device__ __forceinline__ void row_stats(const float* row, float& mu, float& rs) {
  float s = 0.f, s2 = 0.f;
  for (int k = 0; k < DIM; ++k) { float v = row[k]; s += v; s2 += v * v; }
  mu = s * (1.0f / DIM);
  float var = s2 * (1.0f / DIM) - mu * mu;
  rs = rsqrtf(var + 1e-5f);
}

// ---------------- CDNA5 WMMA fragment loaders ----------------
// A (16x32 MxK, bf16): lane = half*16 + row; VGPR j<4 -> K = 8*half + 2j,
// j>=4 -> K = 16 + 8*half + 2(j-4). Row-major source, ld in elements.
__device__ __forceinline__ FragAB load_frag_a(const uint16_t* base, int ld, int lane) {
  FragAB f;
  const int row = lane & 15;
  const int h8  = (lane >> 4) << 3;
  const uint16_t* p = base + row * ld;
#pragma unroll
  for (int j = 0; j < 4; ++j) f.u[j]     = *(const uint32_t*)(p + h8 + 2 * j);
#pragma unroll
  for (int j = 0; j < 4; ++j) f.u[4 + j] = *(const uint32_t*)(p + 16 + h8 + 2 * j);
  return f;
}
// B (32x16 KxN, bf16), loaded from B^T stored row-major [N][K]:
// lane = half*16 + col; VGPR j<4 -> K = 8*half + 2j, j>=4 -> K = 16 + 8*half + 2(j-4).
__device__ __forceinline__ FragAB load_frag_b(const uint16_t* baseT, int ld, int lane) {
  FragAB f;
  const int col = lane & 15;
  const int h8  = (lane >> 4) << 3;
  const uint16_t* p = baseT + col * ld;
#pragma unroll
  for (int j = 0; j < 4; ++j) f.u[j]     = *(const uint32_t*)(p + h8 + 2 * j);
#pragma unroll
  for (int j = 0; j < 4; ++j) f.u[4 + j] = *(const uint32_t*)(p + 16 + h8 + 2 * j);
  return f;
}

// Workgroup GEMM: C(rows x ncols) (+)= A(rows x kdim) * W^T(ncols x kdim)^T.
// A,Wt are bf16(u16) row-major in LDS/global; C is f32 row-major in LDS.
// 4 waves, tiles round-robined across waves.
__device__ __forceinline__ void tile_gemm(const uint16_t* __restrict__ As, int lda,
                                          const uint16_t* __restrict__ Wt, int ldw,
                                          float* __restrict__ Cs, int ldc,
                                          int rows, int ncols, int kdim,
                                          int tid, bool accum) {
  const int wave = tid >> 5;
  const int lane = tid & 31;
  const int mt = rows >> 4, nt = ncols >> 4, nk = kdim >> 5;
  for (int t = wave; t < mt * nt; t += 4) {
    const int tm = t / nt, tn = t % nt;
    FragC acc;
#pragma unroll
    for (int v = 0; v < 8; ++v) acc.f[v] = 0.0f;
    for (int kk = 0; kk < nk; ++kk) {
      FragAB fa = load_frag_a(As + (tm * 16) * lda + kk * 32, lda, lane);
      FragAB fb = load_frag_b(Wt + (tn * 16) * ldw + kk * 32, ldw, lane);
      acc.v = __builtin_amdgcn_wmma_f32_16x16x32_bf16(false, fa.v, false, fb.v,
                                                      (short)0, acc.v, false, false);
    }
    const int n  = lane & 15;
    const int mb = (lane >> 4) << 3;
    float* cp = Cs + (tm * 16 + mb) * ldc + tn * 16 + n;
#pragma unroll
    for (int v = 0; v < 8; ++v) {
      if (accum) cp[v * ldc] += acc.f[v];
      else       cp[v * ldc]  = acc.f[v];
    }
  }
}

__device__ __forceinline__ void cp_u16(uint16_t* dst, const uint16_t* src, int n, int tid) {
  const uint32_t* s = (const uint32_t*)src;
  uint32_t* d = (uint32_t*)dst;
  const int n2 = n >> 1;
  for (int i = tid; i < n2; i += NTHREADS) d[i] = s[i];
}
__device__ __forceinline__ void cp_u16_strided(uint16_t* dst, const uint16_t* src,
                                               int nrows, int ncols, int sld, int tid) {
  const int n = nrows * ncols;
  for (int e = tid; e < n; e += NTHREADS) {
    int r = e / ncols, c = e % ncols;
    dst[e] = src[r * sld + c];
  }
}

// ---------------- Tensor Data Mover staging (CDNA5) ----------------
// Stage a 2D bf16 tile (nrows x row_u16, source leading dim src_ld_u16) from
// global into LDS with a single TDM descriptor, issued by wave 0 and tracked
// with TENSORcnt. Data is moved in 8-byte units; D# per CDNA5 ISA.
#if defined(__has_builtin)
#if __has_builtin(__builtin_amdgcn_tensor_load_to_lds) && __has_builtin(__builtin_amdgcn_s_wait_tensorcnt)
#define USE_TDM 1
#endif
#endif
#ifndef USE_TDM
#define USE_TDM 0
#endif

#if USE_TDM
typedef __attribute__((ext_vector_type(4))) unsigned int u32x4;
typedef __attribute__((ext_vector_type(8))) int         i32x8;
typedef __attribute__((ext_vector_type(4))) int         i32x4;

__device__ __forceinline__ void tdm_stage(uint16_t* dst, const uint16_t* src,
                                          int row_u16, int nrows, int src_ld_u16, int tid) {
  if (tid < 32) {
    const unsigned lds = (unsigned)(uintptr_t)dst;                  // LDS byte offset
    const unsigned long long ga = (unsigned long long)(uintptr_t)src;
    const unsigned tile0   = (unsigned)(row_u16 >> 2);              // 8B units per row
    const unsigned stride0 = (unsigned)(src_ld_u16 >> 2);           // 8B units
    u32x4 g0;
    g0[0] = 1u;                                                     // count=1 (valid), user D#
    g0[1] = lds;                                                    // lds_addr (bytes)
    g0[2] = (unsigned)(ga & 0xFFFFFFFFu);                           // global_addr[31:0]
    g0[3] = (unsigned)((ga >> 32) & 0x01FFFFFFu) | (2u << 30);      // addr[56:32] | type=2
    i32x8 g1;
    g1[0] = (int)(3u << 16);                                        // data_size = 8 bytes
    g1[1] = 0;                                                      // no atomic barrier
    g1[2] = (int)0x4000u;                                           // tensor_dim0 = 1<<30 (hi16)
    g1[3] = (int)(0x4000u | (tile0 << 16));                         // tensor_dim1 hi | tile_dim0
    g1[4] = (int)(nrows & 0xFFFF);                                  // tile_dim1 (tile_dim2 = 0)
    g1[5] = (int)stride0;                                           // tensor_dim0_stride[31:0]
    g1[6] = 0;
    g1[7] = 0;
    i32x4 z4;
    z4[0] = 0; z4[1] = 0; z4[2] = 0; z4[3] = 0;
    i32x8 z8;
    z8[0] = 0; z8[1] = 0; z8[2] = 0; z8[3] = 0;
    z8[4] = 0; z8[5] = 0; z8[6] = 0; z8[7] = 0;
    __builtin_amdgcn_tensor_load_to_lds(g0, g1, z4, z4, z8, 0);     // 6-arg (clang-23) form
    __builtin_amdgcn_s_wait_tensorcnt(0);
  }
}
#endif

__device__ __forceinline__ void stage_w(uint16_t* dst, const uint16_t* src,
                                        int row_u16, int nrows, int src_ld_u16, int tid) {
#if USE_TDM
  tdm_stage(dst, src, row_u16, nrows, src_ld_u16, tid);
#else
  if (src_ld_u16 == row_u16) cp_u16(dst, src, row_u16 * nrows, tid);
  else cp_u16_strided(dst, src, nrows, row_u16, src_ld_u16, tid);
#endif
}

// ---------------- small utility kernels ----------------
__global__ void k_wconv(const float* __restrict__ src, uint16_t* __restrict__ dst, int K, int N) {
  int i = blockIdx.x * 256 + threadIdx.x;
  if (i < K * N) {
    int k = i / N, n = i % N;
    dst[(size_t)n * K + k] = f2bf(src[(size_t)k * N + n]);   // store W^T in bf16
  }
}
__global__ void k_zero(float* __restrict__ p, long n) {
  long i = (long)blockIdx.x * 256 + threadIdx.x;
  if (i < n) p[i] = 0.f;
}
__global__ void k_copy(const float* __restrict__ s, float* __restrict__ d, long n) {
  long i = (long)blockIdx.x * 256 + threadIdx.x;
  if (i < n) d[i] = s[i];
}
__global__ void k_repc(const float* __restrict__ c, float* __restrict__ d, long n) {
  long i = (long)blockIdx.x * 256 + threadIdx.x;
  if (i < n) d[i] = c[i % ((long)M_ATOMS * DIM)];
}
__global__ void k_counts(const int* __restrict__ idx, float* __restrict__ cnt) {
  int m = blockIdx.x * 256 + threadIdx.x;
  if (m < M_ATOMS) atomicAdd(&cnt[idx[m]], 1.0f);
}
// q = repeat(q,T) + r @ rq_w
__global__ void k_prep(const float* __restrict__ q, const float* __restrict__ r,
                       const float* __restrict__ rqw, float* __restrict__ wA) {
  size_t i = (size_t)blockIdx.x * 256 + threadIdx.x;
  const size_t n = (size_t)BT * M_ATOMS * DIM;
  if (i >= n) return;
  size_t tok = i / DIM; int d = (int)(i % DIM);
  size_t m = tok % M_ATOMS;
  const float* rr = r + tok * 3;
  wA[i] = q[m * DIM + d] + rr[0] * rqw[d] + rr[1] * rqw[DIM + d] + rr[2] * rqw[2 * DIM + d];
}

// ---------------- layer kernel 1: AdaLN1 + q/k/v/gate projections ----------------
__global__ void __launch_bounds__(NTHREADS)
k_adaln1_proj(const float* __restrict__ wA, const float* __restrict__ c_in,
              const float* __restrict__ sn_w, const float* __restrict__ sc_b,
              const float* __restrict__ bq,
              const uint16_t* __restrict__ t_sc, const uint16_t* __restrict__ t_bi,
              const uint16_t* __restrict__ t_wq, const uint16_t* __restrict__ t_wk,
              const uint16_t* __restrict__ t_wv, const uint16_t* __restrict__ t_wg,
              uint16_t* __restrict__ wKb, uint16_t* __restrict__ wVb,
              uint16_t* __restrict__ wQHb, uint16_t* __restrict__ wGb) {
  __shared__ __align__(16) uint16_t s_w[DIM * DIM];
  __shared__ __align__(16) uint16_t s_sn[WIN * DIM];
  __shared__ __align__(16) uint16_t s_b[WIN * DIM];
  __shared__ float s_lnA[WIN * DIM];
  __shared__ float s_x[WIN * DIM];
  __shared__ float s_y[WIN * DIM];
  __shared__ float s_mu[WIN], s_rs[WIN], s_mu2[WIN], s_rs2[WIN];

  const int tid = threadIdx.x;
  const int g = blockIdx.x;
  const int m0 = (g % NWIN) * WIN;
  const size_t base = (size_t)g * WIN;
  const float* qrows = wA + base * DIM;
  const float* crows = c_in + (size_t)m0 * DIM;

  if (tid < WIN) row_stats(qrows + (size_t)tid * DIM, s_mu[tid], s_rs[tid]);
  else if (tid >= 64 && tid < 64 + WIN) {
    int r = tid - 64;
    row_stats(crows + (size_t)r * DIM, s_mu2[r], s_rs2[r]);
  }
  if (tid == 0) __builtin_prefetch(t_sc, 0, 3);
  __syncthreads();
  for (int e = tid; e < WIN * DIM; e += NTHREADS) {
    int r = e / DIM, k = e % DIM;
    s_lnA[e] = (qrows[(size_t)r * DIM + k] - s_mu[r]) * s_rs[r];
    s_sn[e] = f2bf((crows[(size_t)r * DIM + k] - s_mu2[r]) * s_rs2[r] * sn_w[k]);
  }
  __syncthreads();
  stage_w(s_w, t_sc, DIM, DIM, DIM, tid); __syncthreads();
  tile_gemm(s_sn, DIM, s_w, DIM, s_x, DIM, WIN, DIM, DIM, tid, false); __syncthreads();
  stage_w(s_w, t_bi, DIM, DIM, DIM, tid); __syncthreads();
  tile_gemm(s_sn, DIM, s_w, DIM, s_y, DIM, WIN, DIM, DIM, tid, false); __syncthreads();
  for (int e = tid; e < WIN * DIM; e += NTHREADS) {
    int k = e % DIM;
    s_b[e] = f2bf(sigm(s_x[e] + sc_b[k]) * s_lnA[e] + s_y[e]);
  }
  __syncthreads();
  // K projection (project-then-gather: done once per token)
  stage_w(s_w, t_wk, DIM, DIM, DIM, tid); __syncthreads();
  tile_gemm(s_b, DIM, s_w, DIM, s_x, DIM, WIN, DIM, DIM, tid, false); __syncthreads();
  for (int e = tid; e < WIN * DIM; e += NTHREADS) wKb[base * DIM + e] = f2bf(s_x[e]);
  __syncthreads();
  // V projection
  stage_w(s_w, t_wv, DIM, DIM, DIM, tid); __syncthreads();
  tile_gemm(s_b, DIM, s_w, DIM, s_x, DIM, WIN, DIM, DIM, tid, false); __syncthreads();
  for (int e = tid; e < WIN * DIM; e += NTHREADS) wVb[base * DIM + e] = f2bf(s_x[e]);
  __syncthreads();
  // Q projection (+ bias)
  stage_w(s_w, t_wq, DIM, DIM, DIM, tid); __syncthreads();
  tile_gemm(s_b, DIM, s_w, DIM, s_x, DIM, WIN, DIM, DIM, tid, false); __syncthreads();
  for (int e = tid; e < WIN * DIM; e += NTHREADS) wQHb[base * DIM + e] = f2bf(s_x[e] + bq[e % DIM]);
  __syncthreads();
  // attention gate = sigmoid(b @ wg)
  stage_w(s_w, t_wg, DIM, DIM, DIM, tid); __syncthreads();
  tile_gemm(s_b, DIM, s_w, DIM, s_x, DIM, WIN, DIM, DIM, tid, false); __syncthreads();
  for (int e = tid; e < WIN * DIM; e += NTHREADS) wGb[base * DIM + e] = f2bf(sigm(s_x[e]));
}

// ---------------- layer kernel 2: windowed pair-bias attention ----------------
__global__ void __launch_bounds__(NTHREADS)
k_attn(const uint16_t* __restrict__ wQHb, const uint16_t* __restrict__ wKb,
       const uint16_t* __restrict__ wVb, const uint16_t* __restrict__ wGb,
       const float* __restrict__ aeb, const float* __restrict__ mask_in,
       uint16_t* __restrict__ wGOb, int layer) {
  __shared__ __align__(16) uint16_t s_qh[WIN * DIM];
  __shared__ __align__(16) uint16_t s_k[HKEYS * DIM];
  __shared__ __align__(16) uint16_t s_vt[DIM * HKEYS];   // V^T : [d][key]
  __shared__ float s_log[WIN * HKEYS];
  __shared__ __align__(16) uint16_t s_p[WIN * HKEYS];
  __shared__ float s_o[WIN * DIM];

  const int tid = threadIdx.x;
  const int g = blockIdx.x;
  const int b = g / NWIN, wdx = g % NWIN;
  const size_t qbase = (size_t)g * WIN;
  const int a0 = wdx * WIN - OFFK;

  stage_w(s_qh, wQHb + qbase * DIM, DIM, WIN, DIM, tid);
  for (int e = tid; e < HKEYS * DIM; e += NTHREADS) {
    int j = e / DIM, d = e % DIM;
    int ak = a0 + j;
    uint16_t kv = 0, vv = 0;
    if (ak >= 0 && ak < M_ATOMS) {
      size_t row = (size_t)b * M_ATOMS + ak;
      kv = wKb[row * DIM + d];
      vv = wVb[row * DIM + d];
    }
    s_k[j * DIM + d] = kv;
    s_vt[d * HKEYS + j] = vv;
  }
  __syncthreads();

  for (int h = 0; h < NHEADS; ++h) {
    // logits = Q_h (32x32) @ K_h^T (32x128): one WMMA k-step per tile
    tile_gemm(s_qh + h * DHEAD, DIM, s_k + h * DHEAD, DIM, s_log, HKEYS,
              WIN, HKEYS, DHEAD, tid, false);
    __syncthreads();
    if (tid < WIN) {
      const int qv = tid;
      const float* bp = aeb + ((size_t)(wdx * WIN + qv) * HKEYS) * (LAYERS * NHEADS)
                            + (size_t)layer * NHEADS + h;
      float mx = -1e30f;
      for (int j = 0; j < HKEYS; ++j) {
        int ak = a0 + j;
        float mval = (ak >= 0 && ak < M_ATOMS) ? mask_in[ak] : 0.0f;
        float v = s_log[qv * HKEYS + j] * INV_SQRT_DH + bp[(size_t)j * (LAYERS * NHEADS)];
        if (mval <= 0.5f) v = -1e9f;
        s_log[qv * HKEYS + j] = v;
        mx = fmaxf(mx, v);
      }
      float sum = 0.f;
      for (int j = 0; j < HKEYS; ++j) {
        float ev = __expf(s_log[qv * HKEYS + j] - mx);
        sum += ev;
        s_log[qv * HKEYS + j] = ev;
      }
      float rs = 1.0f / sum;
      for (int j = 0; j < HKEYS; ++j) s_p[qv * HKEYS + j] = f2bf(s_log[qv * HKEYS + j] * rs);
    }
    __syncthreads();
    // O_h = P (32x128) @ V_h (128x32)
    tile_gemm(s_p, HKEYS, s_vt + (size_t)(h * DHEAD) * HKEYS, HKEYS,
              s_o + h * DHEAD, DIM, WIN, DHEAD, HKEYS, tid, false);
    __syncthreads();
  }
  for (int e = tid; e < WIN * DIM; e += NTHREADS) {
    float gate = bf2f(wGb[qbase * DIM + e]);
    wGOb[qbase * DIM + e] = f2bf(gate * s_o[e]);
  }
}

// ---------------- layer kernel 3: output proj + AdaLN2 + conditioned transition ----------------
__global__ void __launch_bounds__(NTHREADS)
k_post(float* __restrict__ wA, const uint16_t* __restrict__ wGOb,
       const float* __restrict__ c_in,
       const float* __restrict__ opb, const float* __restrict__ sn2_w,
       const float* __restrict__ sc2_b, const float* __restrict__ tob,
       const uint16_t* __restrict__ t_wo, const uint16_t* __restrict__ t_op,
       const uint16_t* __restrict__ t_sc2, const uint16_t* __restrict__ t_bi2,
       const uint16_t* __restrict__ t_gw, const uint16_t* __restrict__ t_ab,
       const uint16_t* __restrict__ t_ba, const uint16_t* __restrict__ t_tow) {
  __shared__ __align__(16) uint16_t s_w[DIM * DIM];
  __shared__ __align__(16) uint16_t s_cb[WIN * DIM];
  __shared__ __align__(16) uint16_t s_sn[WIN * DIM];
  __shared__ __align__(16) uint16_t s_t[WIN * DIM];
  __shared__ __align__(16) uint16_t s_go[WIN * DIM];
  __shared__ float s_x[WIN * DIM], s_y[WIN * DIM], s_z[WIN * DIM], s_a[WIN * DIM];
  __shared__ __align__(16) uint16_t s_ab[WIN * HID2];
  __shared__ __align__(16) uint16_t s_bb[WIN * HID2];
  __shared__ float s_mu[WIN], s_rs[WIN];

  const int tid = threadIdx.x;
  const int g = blockIdx.x;
  const int m0 = (g % NWIN) * WIN;
  const size_t base = (size_t)g * WIN;
  const float* crows = c_in + (size_t)m0 * DIM;
  float* arows = wA + base * DIM;

  for (int e = tid; e < WIN * DIM; e += NTHREADS) s_cb[e] = f2bf(crows[e]);
  stage_w(s_go, wGOb + base * DIM, DIM, WIN, DIM, tid);
  __syncthreads();
  // x = (gate*o) @ wo
  stage_w(s_w, t_wo, DIM, DIM, DIM, tid); __syncthreads();
  tile_gemm(s_go, DIM, s_w, DIM, s_x, DIM, WIN, DIM, DIM, tid, false); __syncthreads();
  // y = c @ outproj_w
  stage_w(s_w, t_op, DIM, DIM, DIM, tid); __syncthreads();
  tile_gemm(s_cb, DIM, s_w, DIM, s_y, DIM, WIN, DIM, DIM, tid, false); __syncthreads();
  for (int e = tid; e < WIN * DIM; e += NTHREADS) {
    int k = e % DIM;
    s_a[e] = arows[e] + sigm(s_y[e] + opb[k]) * s_x[e];
  }
  __syncthreads();
  // AdaLN2: ln(a) and ln(c)*sn2
  if (tid < WIN) row_stats(&s_a[tid * DIM], s_mu[tid], s_rs[tid]);
  __syncthreads();
  for (int e = tid; e < WIN * DIM; e += NTHREADS) {
    int r = e / DIM;
    s_x[e] = (s_a[e] - s_mu[r]) * s_rs[r];
  }
  __syncthreads();
  if (tid < WIN) row_stats(crows + (size_t)tid * DIM, s_mu[tid], s_rs[tid]);
  __syncthreads();
  for (int e = tid; e < WIN * DIM; e += NTHREADS) {
    int r = e / DIM, k = e % DIM;
    s_sn[e] = f2bf((crows[(size_t)r * DIM + k] - s_mu[r]) * s_rs[r] * sn2_w[k]);
  }
  __syncthreads();
  stage_w(s_w, t_sc2, DIM, DIM, DIM, tid); __syncthreads();
  tile_gemm(s_sn, DIM, s_w, DIM, s_y, DIM, WIN, DIM, DIM, tid, false); __syncthreads();
  stage_w(s_w, t_bi2, DIM, DIM, DIM, tid); __syncthreads();
  tile_gemm(s_sn, DIM, s_w, DIM, s_z, DIM, WIN, DIM, DIM, tid, false); __syncthreads();
  for (int e = tid; e < WIN * DIM; e += NTHREADS) {
    int k = e % DIM;
    s_t[e] = f2bf(sigm(s_y[e] + sc2_b[k]) * s_x[e] + s_z[e]);
  }
  __syncthreads();
  // ab = t @ ab_w  (256 cols, chunks of 128)
  for (int ch = 0; ch < 2; ++ch) {
    stage_w(s_w, t_ab + (size_t)ch * DIM * DIM, DIM, DIM, DIM, tid); __syncthreads();
    tile_gemm(s_t, DIM, s_w, DIM, s_x, DIM, WIN, DIM, DIM, tid, false); __syncthreads();
    for (int e = tid; e < WIN * DIM; e += NTHREADS) {
      int r = e / DIM, cc = e % DIM;
      s_ab[r * HID2 + ch * DIM + cc] = f2bf(s_x[e]);
    }
    __syncthreads();
  }
  // bb = silu(x1) * x2 * ab  (x1 = cols 0..255, x2 = cols 256..511 of t @ gate_w)
  for (int ch = 0; ch < 2; ++ch) {
    stage_w(s_w, t_gw + (size_t)(ch * DIM) * DIM, DIM, DIM, DIM, tid); __syncthreads();
    tile_gemm(s_t, DIM, s_w, DIM, s_y, DIM, WIN, DIM, DIM, tid, false); __syncthreads();
    stage_w(s_w, t_gw + (size_t)(HID2 + ch * DIM) * DIM, DIM, DIM, DIM, tid); __syncthreads();
    tile_gemm(s_t, DIM, s_w, DIM, s_z, DIM, WIN, DIM, DIM, tid, false); __syncthreads();
    for (int e = tid; e < WIN * DIM; e += NTHREADS) {
      int r = e / DIM, cc = e % DIM;
      float x1 = s_y[e];
      float bbv = (x1 * sigm(x1)) * s_z[e] * bf2f(s_ab[r * HID2 + ch * DIM + cc]);
      s_bb[r * HID2 + ch * DIM + cc] = f2bf(bbv);
    }
    __syncthreads();
  }
  // y2 = bb @ ba_w  (K = 256, two K-chunks accumulated; TDM does the strided gather)
  for (int kk = 0; kk < 2; ++kk) {
    stage_w(s_w, t_ba + kk * DIM, DIM, DIM, HID2, tid); __syncthreads();
    tile_gemm(s_bb + kk * DIM, HID2, s_w, DIM, s_x, DIM, WIN, DIM, DIM, tid, kk != 0);
    __syncthreads();
  }
  // a += sigmoid(c @ trans_out_w + b) * y2
  stage_w(s_w, t_tow, DIM, DIM, DIM, tid); __syncthreads();
  tile_gemm(s_cb, DIM, s_w, DIM, s_y, DIM, WIN, DIM, DIM, tid, false); __syncthreads();
  for (int e = tid; e < WIN * DIM; e += NTHREADS) {
    int k = e % DIM;
    arows[e] = s_a[e] + sigm(s_y[e] + tob[k]) * s_x[e];
  }
}

// ---------------- layer kernel 4: temporal attention with decay (2 atoms / WG) ----------------
__global__ void __launch_bounds__(NTHREADS)
k_temporal(float* __restrict__ wA, const float* __restrict__ ts,
           const float* __restrict__ ln_w, const float* __restrict__ ln_b,
           const uint16_t* __restrict__ t_qkv, const uint16_t* __restrict__ t_two,
           const float* __restrict__ dec) {
  __shared__ float s_x[32 * DIM];
  __shared__ __align__(16) uint16_t s_h[32 * DIM];
  __shared__ __align__(16) uint16_t s_w[DIM * DIM];
  __shared__ __align__(16) uint16_t s_qb[32 * DIM];
  __shared__ __align__(16) uint16_t s_kb[32 * DIM];
  __shared__ float s_f[32 * DIM];
  __shared__ __align__(16) uint16_t s_vt[2 * DIM * 32];  // [atom][d][kf padded to 32]
  __shared__ float s_lg[2 * NHEADS * 16 * 16];
  __shared__ __align__(16) uint16_t s_p[2 * NHEADS * 16 * 32];
  __shared__ float s_o[32 * DIM];
  __shared__ __align__(16) uint16_t s_ob[32 * DIM];
  __shared__ float s_mu[32], s_rs[32], s_ts[T_FRAMES], s_dec[NHEADS];

  const int tid = threadIdx.x;
  const int wave = tid >> 5, lane = tid & 31;
  const int atom0 = blockIdx.x * 2;

  if (tid < T_FRAMES) s_ts[tid] = ts[tid];
  if (tid >= 32 && tid < 32 + NHEADS) s_dec[tid - 32] = softplusf(dec[tid - 32]);

  for (int e = tid; e < 32 * DIM; e += NTHREADS) {
    int r = e / DIM, d = e % DIM;
    int aL = r >> 4, fr = r & 15;
    s_x[e] = wA[((size_t)fr * M_ATOMS + atom0 + aL) * DIM + d];
  }
  __syncthreads();
  if (tid < 32) row_stats(&s_x[tid * DIM], s_mu[tid], s_rs[tid]);
  __syncthreads();
  for (int e = tid; e < 32 * DIM; e += NTHREADS) {
    int r = e / DIM, d = e % DIM;
    s_h[e] = f2bf((s_x[e] - s_mu[r]) * s_rs[r] * ln_w[d] + ln_b[d]);
  }
  __syncthreads();
  // q / k / v projections
  for (int seg = 0; seg < 3; ++seg) {
    stage_w(s_w, t_qkv + (size_t)seg * DIM * DIM, DIM, DIM, DIM, tid);
    __syncthreads();
    tile_gemm(s_h, DIM, s_w, DIM, s_f, DIM, 32, DIM, DIM, tid, false);
    __syncthreads();
    if (seg == 0) {
      for (int e = tid; e < 32 * DIM; e += NTHREADS) s_qb[e] = f2bf(s_f[e]);
    } else if (seg == 1) {
      for (int e = tid; e < 32 * DIM; e += NTHREADS) s_kb[e] = f2bf(s_f[e]);
    } else {
      for (int e = tid; e < 2 * DIM * 32; e += NTHREADS) {
        int a = e / (DIM * 32);
        int rem = e % (DIM * 32);
        int d = rem / 32, kf = rem % 32;
        s_vt[e] = (kf < T_FRAMES) ? f2bf(s_f[(a * 16 + kf) * DIM + d]) : (uint16_t)0;
      }
    }
    __syncthreads();
  }
  // logits: one 16x16x32 WMMA per (atom, head)
  for (int task = wave; task < 2 * NHEADS; task += 4) {
    int a = task >> 2, h = task & 3;
    FragC acc;
#pragma unroll
    for (int v = 0; v < 8; ++v) acc.f[v] = 0.f;
    FragAB fa = load_frag_a(s_qb + (a * 16) * DIM + h * DHEAD, DIM, lane);
    FragAB fb = load_frag_b(s_kb + (a * 16) * DIM + h * DHEAD, DIM, lane);
    acc.v = __builtin_amdgcn_wmma_f32_16x16x32_bf16(false, fa.v, false, fb.v,
                                                    (short)0, acc.v, false, false);
    int n = lane & 15, mb = (lane >> 4) << 3;
#pragma unroll
    for (int v = 0; v < 8; ++v) s_lg[task * 256 + (mb + v) * 16 + n] = acc.f[v];
  }
  __syncthreads();
  // softmax with decay bias (one thread per (atom, head, query frame))
  {
    int task = tid >> 4, qf = tid & 15;
    int h = task & 3;
    float dsp = s_dec[h];
    float tq = s_ts[qf];
    float vals[T_FRAMES];
    float mx = -1e30f;
    for (int kf = 0; kf < T_FRAMES; ++kf) {
      float v = s_lg[task * 256 + qf * 16 + kf] * INV_SQRT_DH
              - dsp * fabsf(tq - s_ts[kf]) * 1e-9f;
      vals[kf] = v; mx = fmaxf(mx, v);
    }
    float sum = 0.f;
    for (int kf = 0; kf < T_FRAMES; ++kf) { vals[kf] = __expf(vals[kf] - mx); sum += vals[kf]; }
    float rsum = 1.0f / sum;
    uint16_t* pr = s_p + (task * 16 + qf) * 32;
    for (int kf = 0; kf < T_FRAMES; ++kf) pr[kf] = f2bf(vals[kf] * rsum);
    for (int kf = T_FRAMES; kf < 32; ++kf) pr[kf] = 0;
  }
  __syncthreads();
  // O = P @ V : two 16-col tiles per (atom, head), K padded to 32
  for (int t16 = wave; t16 < 16; t16 += 4) {
    int a = t16 >> 3, h = (t16 >> 1) & 3, ntile = t16 & 1;
    FragC acc;
#pragma unroll
    for (int v = 0; v < 8; ++v) acc.f[v] = 0.f;
    FragAB fa = load_frag_a(s_p + ((a * NHEADS + h) * 16) * 32, 32, lane);
    FragAB fb = load_frag_b(s_vt + (a * DIM + h * DHEAD + ntile * 16) * 32, 32, lane);
    acc.v = __builtin_amdgcn_wmma_f32_16x16x32_bf16(false, fa.v, false, fb.v,
                                                    (short)0, acc.v, false, false);
    int n = lane & 15, mb = (lane >> 4) << 3;
#pragma unroll
    for (int v = 0; v < 8; ++v)
      s_o[(a * 16 + mb + v) * DIM + h * DHEAD + ntile * 16 + n] = acc.f[v];
  }
  __syncthreads();
  for (int e = tid; e < 32 * DIM; e += NTHREADS) s_ob[e] = f2bf(s_o[e]);
  __syncthreads();
  stage_w(s_w, t_two, DIM, DIM, DIM, tid);
  __syncthreads();
  tile_gemm(s_ob, DIM, s_w, DIM, s_f, DIM, 32, DIM, DIM, tid, false);
  __syncthreads();
  for (int e = tid; e < 32 * DIM; e += NTHREADS) {
    int r = e / DIM, d = e % DIM;
    int aL = r >> 4, fr = r & 15;
    wA[((size_t)fr * M_ATOMS + atom0 + aL) * DIM + d] = s_x[e] + s_f[e];
  }
}

// ---------------- final: ReLU(q @ a2t) + fused segment-mean scatter ----------------
__global__ void __launch_bounds__(NTHREADS)
k_pool(const float* __restrict__ wA, const uint16_t* __restrict__ t_a2t,
       const int* __restrict__ idx, const float* __restrict__ cnt,
       float* __restrict__ out_a) {
  __shared__ __align__(16) uint16_t s_w[DIM * DIM];
  __shared__ __align__(16) uint16_t s_h[WIN * DIM];
  __shared__ float s_y[WIN * DIM];
  __shared__ float s_scale[WIN];
  __shared__ int s_tok[WIN];
  const int tid = threadIdx.x;
  const size_t base = (size_t)blockIdx.x * WIN;
  const int b = (int)(base / M_ATOMS);
  for (int e = tid; e < WIN * DIM; e += NTHREADS) s_h[e] = f2bf(wA[base * DIM + e]);
  if (tid < WIN) {
    int m = (int)((base + tid) % M_ATOMS);
    int tk = idx[m];
    s_tok[tid] = tk;
    s_scale[tid] = 1.0f / (cnt[tk] + 1e-6f);
  }
  __syncthreads();
  for (int ch = 0; ch < TS2 / DIM; ++ch) {
    stage_w(s_w, t_a2t + (size_t)ch * DIM * DIM, DIM, DIM, DIM, tid);
    __syncthreads();
    tile_gemm(s_h, DIM, s_w, DIM, s_y, DIM, WIN, DIM, DIM, tid, false);
    __syncthreads();
    for (int e = tid; e < WIN * DIM; e += NTHREADS) {
      int r = e / DIM, cc = e % DIM;
      float v = s_y[e];
      v = v > 0.f ? v : 0.f;
      atomicAdd(&out_a[((size_t)b * NTOK + s_tok[r]) * TS2 + ch * DIM + cc], v * s_scale[r]);
    }
    __syncthreads();
  }
}

// ---------------- host ----------------
extern "C" void kernel_launch(void* const* d_in, const int* in_sizes, int n_in,
                              void* d_out, int out_size, void* d_ws, size_t ws_size,
                              hipStream_t stream) {
  const float* q_in    = (const float*)d_in[0];
  const float* c_in    = (const float*)d_in[1];
  const float* aeb     = (const float*)d_in[2];
  const float* r_in    = (const float*)d_in[3];
  const float* ts_in   = (const float*)d_in[4];
  const float* mask_in = (const float*)d_in[5];
  const float* rq_w    = (const float*)d_in[6];
  const float* a2t_w   = (const float*)d_in[7];
  const float* a1_sn   = (const float*)d_in[8];
  const float* a1_sc   = (const float*)d_in[9];
  const float* a1_scb  = (const float*)d_in[10];
  const float* a1_bi   = (const float*)d_in[11];
  const float* a2_sn   = (const float*)d_in[12];
  const float* a2_sc   = (const float*)d_in[13];
  const float* a2_scb  = (const float*)d_in[14];
  const float* a2_bi   = (const float*)d_in[15];
  const float* wq_w    = (const float*)d_in[16];
  const float* bq_b    = (const float*)d_in[17];
  const float* wk_w    = (const float*)d_in[18];
  const float* wv_w    = (const float*)d_in[19];
  const float* wg_w    = (const float*)d_in[20];
  const float* wo_w    = (const float*)d_in[21];
  const float* op_w    = (const float*)d_in[22];
  const float* op_b    = (const float*)d_in[23];
  const float* gw_w    = (const float*)d_in[24];
  const float* ab_w    = (const float*)d_in[25];
  const float* ba_w    = (const float*)d_in[26];
  const float* tow_w   = (const float*)d_in[27];
  const float* tob_b   = (const float*)d_in[28];
  const float* tln_w   = (const float*)d_in[29];
  const float* tln_b   = (const float*)d_in[30];
  const float* qkv_w   = (const float*)d_in[31];
  const float* two_w   = (const float*)d_in[32];
  const float* dec_w   = (const float*)d_in[33];
  const int*   tok_idx = (const int*)d_in[34];

  char* wsp = (char*)d_ws;
  size_t off = 0;
  auto carve = [&](size_t bytes) -> void* {
    off = (off + 255) & ~(size_t)255;
    void* p = (void*)(wsp + off);
    off += bytes;
    return p;
  };
  const size_t NT = (size_t)BT * M_ATOMS;
  float*    wAf  = (float*)carve(NT * DIM * sizeof(float));
  uint16_t* wKb  = (uint16_t*)carve(NT * DIM * 2);
  uint16_t* wVb  = (uint16_t*)carve(NT * DIM * 2);
  uint16_t* wQHb = (uint16_t*)carve(NT * DIM * 2);
  uint16_t* wGb  = (uint16_t*)carve(NT * DIM * 2);
  uint16_t* wGOb = (uint16_t*)carve(NT * DIM * 2);
  float*    wCnt = (float*)carve(NTOK * sizeof(float));

  auto conv = [&](const float* src, int K, int N) -> uint16_t* {
    uint16_t* dst = (uint16_t*)carve((size_t)K * N * 2);
    int n = K * N;
    k_wconv<<<dim3((n + 255) / 256), dim3(256), 0, stream>>>(src, dst, K, N);
    return dst;
  };

  uint16_t *t_sc1[LAYERS], *t_bi1[LAYERS], *t_wq[LAYERS], *t_wk[LAYERS], *t_wv[LAYERS],
           *t_wg[LAYERS], *t_wo[LAYERS], *t_op[LAYERS], *t_sc2[LAYERS], *t_bi2[LAYERS],
           *t_gw[LAYERS], *t_ab[LAYERS], *t_ba[LAYERS], *t_tow[LAYERS],
           *t_qkv[LAYERS], *t_two[LAYERS];
  for (int i = 0; i < LAYERS; ++i) {
    const size_t dd = (size_t)DIM * DIM;
    t_sc1[i] = conv(a1_sc + i * dd, DIM, DIM);
    t_bi1[i] = conv(a1_bi + i * dd, DIM, DIM);
    t_wq[i]  = conv(wq_w + i * dd, DIM, DIM);
    t_wk[i]  = conv(wk_w + i * dd, DIM, DIM);
    t_wv[i]  = conv(wv_w + i * dd, DIM, DIM);
    t_wg[i]  = conv(wg_w + i * dd, DIM, DIM);
    t_wo[i]  = conv(wo_w + i * dd, DIM, DIM);
    t_op[i]  = conv(op_w + i * dd, DIM, DIM);
    t_sc2[i] = conv(a2_sc + i * dd, DIM, DIM);
    t_bi2[i] = conv(a2_bi + i * dd, DIM, DIM);
    t_gw[i]  = conv(gw_w + (size_t)i * DIM * 2 * HID2, DIM, 2 * HID2);
    t_ab[i]  = conv(ab_w + (size_t)i * DIM * HID2, DIM, HID2);
    t_ba[i]  = conv(ba_w + (size_t)i * HID2 * DIM, HID2, DIM);
    t_tow[i] = conv(tow_w + i * dd, DIM, DIM);
    t_qkv[i] = conv(qkv_w + (size_t)i * DIM * 3 * DIM, DIM, 3 * DIM);
    t_two[i] = conv(two_w + i * dd, DIM, DIM);
  }
  uint16_t* t_a2t = conv(a2t_w, DIM, TS2);

  {
    size_t n = NT * DIM;
    k_prep<<<dim3((unsigned)((n + 255) / 256)), dim3(256), 0, stream>>>(q_in, r_in, rq_w, wAf);
  }
  for (int i = 0; i < LAYERS; ++i) {
    k_adaln1_proj<<<dim3(BT * NWIN), dim3(NTHREADS), 0, stream>>>(
        wAf, c_in, a1_sn + i * DIM, a1_scb + i * DIM, bq_b + i * DIM,
        t_sc1[i], t_bi1[i], t_wq[i], t_wk[i], t_wv[i], t_wg[i],
        wKb, wVb, wQHb, wGb);
    k_attn<<<dim3(BT * NWIN), dim3(NTHREADS), 0, stream>>>(
        wQHb, wKb, wVb, wGb, aeb, mask_in, wGOb, i);
    k_post<<<dim3(BT * NWIN), dim3(NTHREADS), 0, stream>>>(
        wAf, wGOb, c_in, op_b + i * DIM, a2_sn + i * DIM, a2_scb + i * DIM, tob_b + i * DIM,
        t_wo[i], t_op[i], t_sc2[i], t_bi2[i], t_gw[i], t_ab[i], t_ba[i], t_tow[i]);
    k_temporal<<<dim3(M_ATOMS / 2), dim3(NTHREADS), 0, stream>>>(
        wAf, ts_in, tln_w + i * DIM, tln_b + i * DIM, t_qkv[i], t_two[i], dec_w + i * NHEADS);
  }
  float* out_a = (float*)d_out;
  float* out_q = out_a + (size_t)BT * NTOK * TS2;
  float* out_c = out_q + NT * DIM;
  {
    long na = (long)BT * NTOK * TS2;
    k_zero<<<dim3((unsigned)((na + 255) / 256)), dim3(256), 0, stream>>>(out_a, na);
    k_zero<<<dim3((NTOK + 255) / 256), dim3(256), 0, stream>>>(wCnt, (long)NTOK);
  }
  k_counts<<<dim3(M_ATOMS / 256), dim3(256), 0, stream>>>(tok_idx, wCnt);
  k_pool<<<dim3((unsigned)(NT / WIN)), dim3(NTHREADS), 0, stream>>>(wAf, t_a2t, tok_idx, wCnt, out_a);
  {
    long nq = (long)NT * DIM;
    k_copy<<<dim3((unsigned)((nq + 255) / 256)), dim3(256), 0, stream>>>(wAf, out_q, nq);
    k_repc<<<dim3((unsigned)((nq + 255) / 256)), dim3(256), 0, stream>>>(c_in, out_c, nq);
  }
  (void)in_sizes; (void)n_in; (void)out_size; (void)ws_size;
}
